// Head_26998164423232
// MI455X (gfx1250) — compile-verified, matrix operand order checked by reading, not run on hardware
//
#include <hip/hip_runtime.h>
#include <hip/hip_bf16.h>

typedef __attribute__((ext_vector_type(16))) __bf16 v16bf;
typedef __attribute__((ext_vector_type(8)))  __bf16 v8bf;
typedef __attribute__((ext_vector_type(8)))  float  v8f;
typedef __attribute__((ext_vector_type(4)))  float  v4f;

static constexpr int Cdim = 1024;
static constexpr int HSZ  = 64;
static constexpr int Tdim = 2048;
static constexpr int Bdim = 8;

// Q scale = C^-0.5 (reference) * log2(e): S arrives in log2-domain so the
// softmax can use the raw hardware base-2 exponential (v_exp_f32) with no
// per-element multiply.
static constexpr float QSCALE = 0.03125f * 1.4426950408889634f;

static __device__ __forceinline__ v16bf cat8(v8bf lo, v8bf hi) {
    return __builtin_shufflevector(lo, hi, 0,1,2,3,4,5,6,7,8,9,10,11,12,13,14,15);
}

static __device__ __forceinline__ v8f wmma_bf16(v16bf a, v16bf b, v8f c) {
    // D = A(16x32 bf16) * B(32x16 bf16) + C(16x16 f32)
    return __builtin_amdgcn_wmma_f32_16x16x32_bf16(false, a, false, b, (short)0, c, false, false);
}

// ---- DPP 16-lane butterfly reductions (pure VALU, no LDS/bpermute) --------
template<int CTRL>
static __device__ __forceinline__ float dppf(float v) {
    int r = __builtin_amdgcn_update_dpp(0, __builtin_bit_cast(int, v),
                                        CTRL, 0xF, 0xF, true);
    return __builtin_bit_cast(float, r);
}
static __device__ __forceinline__ float rowmax16(float v) {
    v = fmaxf(v, dppf<0xB1>(v));    // quad_perm [1,0,3,2]  (xor 1)
    v = fmaxf(v, dppf<0x4E>(v));    // quad_perm [2,3,0,1]  (xor 2)
    v = fmaxf(v, dppf<0x124>(v));   // row_ror:4  (fold quads)
    v = fmaxf(v, dppf<0x128>(v));   // row_ror:8
    return v;                        // broadcast over the 16-lane row
}
static __device__ __forceinline__ float rowsum16(float v) {
    v += dppf<0xB1>(v);
    v += dppf<0x4E>(v);
    v += dppf<0x124>(v);
    v += dppf<0x128>(v);
    return v;
}

// ---------------------------------------------------------------------------
// Kernel 1: transpose the three [C, HS] f32 weights into bf16 [HS, C] so that
// WMMA B-fragments become contiguous 16-byte loads along the contraction dim.
// ---------------------------------------------------------------------------
__global__ __launch_bounds__(256) void wt_kernel(const float* __restrict__ Wq,
                                                 const float* __restrict__ Wk,
                                                 const float* __restrict__ Wv,
                                                 __bf16* __restrict__ Wt) {
    int idx = blockIdx.x * 256 + threadIdx.x;      // 3 * 64 * 1024 threads total
    int mat = idx >> 16;
    int rem = idx & 65535;
    int h   = rem >> 10;
    int c   = rem & 1023;
    const float* W = (mat == 0) ? Wq : (mat == 1) ? Wk : Wv;
    Wt[(size_t)mat * 65536 + (h << 10) + c] = (__bf16)W[c * HSZ + h];
}

// ---------------------------------------------------------------------------
// Kernel 2: projections.  KV=false: Q = (x_q @ Wq) * QSCALE  (row-major bf16)
//           KV=true:  K = x_kv @ Wk (row-major), V = x_kv @ Wv stored
//           TRANSPOSED as Vt[b][h][t] (contiguous 16B packs from D-fragment).
// ---------------------------------------------------------------------------
template<bool KV>
__global__ __launch_bounds__(256) void proj_kernel(const float* __restrict__ x,
                                                   const __bf16* __restrict__ WtA,
                                                   const __bf16* __restrict__ WtB,
                                                   __bf16* __restrict__ outA,
                                                   __bf16* __restrict__ outVt,
                                                   float scaleA) {
    const int wave = threadIdx.x >> 5;
    const int lane = threadIdx.x & 31;
    const int ln   = lane & 15;
    const int half = lane >> 4;
    const int hb   = half * 8;
    const int rowBase = blockIdx.x * 128 + wave * 16;

    v8f accA[4] = {};
    v8f accB[4] = {};
    const float* xr = x + (size_t)(rowBase + ln) * Cdim;

    for (int kc = 0; kc < Cdim; kc += 32) {
        v4f f0 = *(const v4f*)(xr + kc + hb);
        v4f f1 = *(const v4f*)(xr + kc + hb + 4);
        v4f f2 = *(const v4f*)(xr + kc + 16 + hb);
        v4f f3 = *(const v4f*)(xr + kc + 16 + hb + 4);
        v16bf a;
        #pragma unroll
        for (int i = 0; i < 4; ++i) {
            a[i]      = (__bf16)f0[i];
            a[4 + i]  = (__bf16)f1[i];
            a[8 + i]  = (__bf16)f2[i];
            a[12 + i] = (__bf16)f3[i];
        }
        #pragma unroll
        for (int nt = 0; nt < 4; ++nt) {
            const __bf16* wp = WtA + (size_t)(nt * 16 + ln) * Cdim + kc;
            v16bf bf = cat8(*(const v8bf*)(wp + hb), *(const v8bf*)(wp + 16 + hb));
            accA[nt] = wmma_bf16(a, bf, accA[nt]);
            if (KV) {
                const __bf16* wp2 = WtB + (size_t)(nt * 16 + ln) * Cdim + kc;
                v16bf bf2 = cat8(*(const v8bf*)(wp2 + hb), *(const v8bf*)(wp2 + 16 + hb));
                accB[nt] = wmma_bf16(a, bf2, accB[nt]);
            }
        }
    }

    #pragma unroll
    for (int nt = 0; nt < 4; ++nt)
        #pragma unroll
        for (int r = 0; r < 8; ++r)
            outA[(size_t)(rowBase + r + 8 * half) * HSZ + nt * 16 + ln] =
                (__bf16)(accA[nt][r] * scaleA);

    if (KV) {
        int bb   = rowBase >> 11;               // rowBase / Tdim
        int tloc = (rowBase & (Tdim - 1)) + 8 * half;
        #pragma unroll
        for (int nt = 0; nt < 4; ++nt) {
            int h = nt * 16 + ln;
            v8bf pk;
            #pragma unroll
            for (int r = 0; r < 8; ++r) pk[r] = (__bf16)accB[nt][r];
            *(v8bf*)(outVt + (((size_t)(bb * HSZ + h)) << 11) + tloc) = pk;
        }
    }
}

// ---------------------------------------------------------------------------
// One 64-key attention tile for one wave (16 query rows).
// MASKED=false: tile entirely below the causal diagonal (no cmp/cndmask).
// All softmax math is in log2 domain (Q pre-scaled by log2e) so every
// exponential is a single v_exp_f32.
// ---------------------------------------------------------------------------
template<bool MASKED>
static __device__ __forceinline__ void attn_tile(
        int k0, int qBase, int ln, int half, int hb,
        v16bf aq0, v16bf aq1,
        const __bf16* __restrict__ Kbb,     // K base for this batch
        const __bf16* __restrict__ Vtb,     // V^T base for this batch
        __bf16 (*__restrict__ ldsP)[64],    // wave-private [16][64] staging
        v8f* __restrict__ o, float* __restrict__ mi, float* __restrict__ li) {
    const float NEGINF = -__builtin_inff();

    // S = Q @ K^T for this 16x64 slab (four 16x16 n-tiles)
    v8f s[4];
    #pragma unroll
    for (int nt = 0; nt < 4; ++nt) {
        const __bf16* kp = Kbb + (size_t)(k0 + nt * 16 + ln) * HSZ;
        v16bf bk0 = cat8(*(const v8bf*)(kp + hb),      *(const v8bf*)(kp + 16 + hb));
        v16bf bk1 = cat8(*(const v8bf*)(kp + 32 + hb), *(const v8bf*)(kp + 48 + hb));
        v8f acc = {};
        acc = wmma_bf16(aq0, bk0, acc);
        acc = wmma_bf16(aq1, bk1, acc);
        s[nt] = acc;
    }

    // Online softmax per row; D layout: row = r + 8*half, col = nt*16 + ln
    #pragma unroll
    for (int r = 0; r < 8; ++r) {
        float sv[4];
        if (MASKED) {
            const int q = qBase + r + 8 * half;
            #pragma unroll
            for (int nt = 0; nt < 4; ++nt)
                sv[nt] = (k0 + nt * 16 + ln <= q) ? s[nt][r] : NEGINF;
        } else {
            #pragma unroll
            for (int nt = 0; nt < 4; ++nt) sv[nt] = s[nt][r];
        }
        float mloc = fmaxf(fmaxf(sv[0], sv[1]), fmaxf(sv[2], sv[3]));
        mloc = rowmax16(mloc);
        const float mnew = fmaxf(mi[r], mloc);
        const float sc = __builtin_amdgcn_exp2f(mi[r] - mnew);
        float p[4];
        #pragma unroll
        for (int nt = 0; nt < 4; ++nt)
            p[nt] = __builtin_amdgcn_exp2f(sv[nt] - mnew);
        float sum = rowsum16((p[0] + p[1]) + (p[2] + p[3]));
        li[r] = li[r] * sc + sum;
        mi[r] = mnew;
        #pragma unroll
        for (int nt = 0; nt < 4; ++nt) o[nt][r] *= sc;
        const int m = r + 8 * half;
        #pragma unroll
        for (int nt = 0; nt < 4; ++nt)
            ldsP[m][nt * 16 + ln] = (__bf16)p[nt];
    }

    // Transpose P (D layout) -> A layout via wave-private LDS.
    // Same-wave LDS ops are in-order; no block barrier needed.
    asm volatile("" ::: "memory");
    v16bf ap0 = cat8(*(const v8bf*)&ldsP[ln][hb],
                     *(const v8bf*)&ldsP[ln][16 + hb]);
    v16bf ap1 = cat8(*(const v8bf*)&ldsP[ln][32 + hb],
                     *(const v8bf*)&ldsP[ln][48 + hb]);
    asm volatile("" ::: "memory");

    // O += P @ V  (V stored transposed: contiguous along keys)
    #pragma unroll
    for (int nt = 0; nt < 4; ++nt) {
        const __bf16* vp = Vtb + (size_t)(nt * 16 + ln) * Tdim + k0;
        v16bf bv0 = cat8(*(const v8bf*)(vp + hb),      *(const v8bf*)(vp + 16 + hb));
        v16bf bv1 = cat8(*(const v8bf*)(vp + 32 + hb), *(const v8bf*)(vp + 48 + hb));
        o[nt] = wmma_bf16(ap0, bv0, o[nt]);
        o[nt] = wmma_bf16(ap1, bv1, o[nt]);
    }
}

// ---------------------------------------------------------------------------
// Kernel 3: causal flash attention, 64-key tiles, mask-free hot loop plus a
// masked diagonal epilogue.  Grid (T/128, B), 8 waves/block, 16 rows/wave.
// ---------------------------------------------------------------------------
__global__ __launch_bounds__(256) void attn_kernel(const __bf16* __restrict__ Qb,
                                                   const __bf16* __restrict__ Kb,
                                                   const __bf16* __restrict__ Vt,
                                                   float* __restrict__ out) {
    __shared__ __align__(16) __bf16 lds[8][16][64];   // per-wave P staging

    const int b    = blockIdx.y;
    const int qt   = blockIdx.x;
    const int wave = threadIdx.x >> 5;
    const int lane = threadIdx.x & 31;
    const int ln   = lane & 15;
    const int half = lane >> 4;
    const int hb   = half * 8;

    const int qBase = qt * 128 + wave * 16;
    const size_t qrow = (size_t)b * Tdim + qBase;
    const __bf16* Kbb = Kb + (size_t)b * Tdim * HSZ;
    const __bf16* Vtb = Vt + (size_t)b * HSZ * Tdim;

    // Q A-fragments for contraction chunks h=[0,32) and h=[32,64)
    const __bf16* qp = Qb + (qrow + ln) * HSZ;
    v16bf aq0 = cat8(*(const v8bf*)(qp + hb),      *(const v8bf*)(qp + 16 + hb));
    v16bf aq1 = cat8(*(const v8bf*)(qp + 32 + hb), *(const v8bf*)(qp + 48 + hb));

    v8f o[4] = {};
    float mi[8], li[8];
    #pragma unroll
    for (int r = 0; r < 8; ++r) { mi[r] = -__builtin_inff(); li[r] = 0.0f; }

    const int nFull = qBase >> 6;        // tiles with k0+63 <= qBase: no mask
    const int nKT   = (qBase + 79) >> 6; // tiles covering keys <= qBase+15

    for (int kt = 0; kt < nFull; ++kt)
        attn_tile<false>(kt * 64, qBase, ln, half, hb, aq0, aq1,
                         Kbb, Vtb, lds[wave], o, mi, li);
    for (int kt = nFull; kt < nKT; ++kt)
        attn_tile<true>(kt * 64, qBase, ln, half, hb, aq0, aq1,
                        Kbb, Vtb, lds[wave], o, mi, li);

    #pragma unroll
    for (int r = 0; r < 8; ++r) {
        const float inv = 1.0f / li[r];
        #pragma unroll
        for (int nt = 0; nt < 4; ++nt)
            out[(qrow + r + 8 * half) * HSZ + nt * 16 + ln] = o[nt][r] * inv;
    }
}

// ---------------------------------------------------------------------------
extern "C" void kernel_launch(void* const* d_in, const int* in_sizes, int n_in,
                              void* d_out, int out_size, void* d_ws, size_t ws_size,
                              hipStream_t stream) {
    const float* x_q  = (const float*)d_in[0];
    const float* x_kv = (const float*)d_in[1];
    const float* Wq   = (const float*)d_in[2];
    const float* Wk   = (const float*)d_in[3];
    const float* Wv   = (const float*)d_in[4];
    float* out = (float*)d_out;

    // Workspace layout (bf16): 3 transposed weights, Q, K, V^T  (~6.4 MB)
    __bf16* Wt = (__bf16*)d_ws;
    const size_t wtElems  = 3ull * HSZ * Cdim;         // 196608
    const size_t qkvElems = (size_t)Bdim * Tdim * HSZ; // 1048576
    __bf16* Qb = Wt + wtElems;
    __bf16* Kb = Qb + qkvElems;
    __bf16* Vt = Kb + qkvElems;

    wt_kernel<<<768, 256, 0, stream>>>(Wq, Wk, Wv, Wt);

    const int rowBlocks = (Bdim * Tdim) / 128;  // 128
    proj_kernel<false><<<rowBlocks, 256, 0, stream>>>(
        x_q, Wt, nullptr, Qb, nullptr, QSCALE /* C^-0.5 * log2e folded into Q */);
    proj_kernel<true><<<rowBlocks, 256, 0, stream>>>(
        x_kv, Wt + (size_t)HSZ * Cdim, Wt + 2ull * HSZ * Cdim, Kb, Vt, 1.0f);

    attn_kernel<<<dim3(Tdim / 128, Bdim), 256, 0, stream>>>(Qb, Kb, Vt, out);
}
